// VGNN_34668976013757
// MI455X (gfx1250) — matrix operand order repaired and checked by reading, not compiled, not run on hardware
//
#include <hip/hip_runtime.h>
#include <stdint.h>

// ---------------------------------------------------------------------------
// Vision-GNN forward for MI455X (gfx1250, wave32, WMMA bf16 -> f32 acc)
// B-panel staged through LDS with CDNA5 async global->LDS copies.
// ---------------------------------------------------------------------------

#define MROWS 32768LL            // 32 batch * 1024 patches
#define NPATCH 1024
#define CDIM 192

typedef __attribute__((ext_vector_type(16))) __bf16 v16bf;
typedef __attribute__((ext_vector_type(8)))  float  v8f;
typedef __attribute__((ext_vector_type(4)))  unsigned int u32x4;
typedef __attribute__((ext_vector_type(8)))  int i32x8;
typedef __attribute__((ext_vector_type(4)))  int i32x4;

union FragBF { v16bf v; uint4 q[2]; };

__device__ __forceinline__ uint16_t f2b(float f) {
  uint32_t u = __float_as_uint(f);
  uint32_t r = (u + 0x7fffu + ((u >> 16) & 1u)) >> 16;   // RNE bf16
  return (uint16_t)r;
}
__device__ __forceinline__ float gelu_exact(float v) {
  return 0.5f * v * (1.0f + erff(v * 0.70710678118654752f));
}
__device__ __forceinline__ v8f zero8() {
  v8f z;
#pragma unroll
  for (int i = 0; i < 8; ++i) z[i] = 0.0f;
  return z;
}

// ---------------------------------------------------------------------------
// Patchify: (32,3,512,512) f32 -> bf16 [32768, 768], feature = c*256+py*16+px
// ---------------------------------------------------------------------------
__global__ __launch_bounds__(256) void k_patchify(const float* __restrict__ x,
                                                  uint16_t* __restrict__ out) {
  long long idx = (long long)blockIdx.x * blockDim.x + threadIdx.x;
  if (idx >= MROWS * 768LL) return;
  int f = (int)(idx % 768);
  long long row = idx / 768;
  int b = (int)(row >> 10);
  int n = (int)(row & 1023);
  int gy = n >> 5, gx = n & 31;
  int c = f >> 8, r = f & 255;
  int py = r >> 4, px = r & 15;
  long long src = (((long long)(b * 3 + c) * 512) + gy * 16 + py) * 512 + gx * 16 + px;
  out[idx] = f2b(x[src]);
}

// ---------------------------------------------------------------------------
// Weight convert: f32 (N,K) row-major -> bf16 (N,Kp) zero-padded
// ---------------------------------------------------------------------------
__global__ __launch_bounds__(256) void k_w2b(const float* __restrict__ w, int N, int K,
                                             int Kp, uint16_t* __restrict__ dst,
                                             long long total) {
  long long idx = (long long)blockIdx.x * blockDim.x + threadIdx.x;
  if (idx >= total) return;
  int kp = (int)(idx % Kp);
  long long n = idx / Kp;
  dst[idx] = (kp < K) ? f2b(w[n * K + kp]) : (uint16_t)0;
}

// ---------------------------------------------------------------------------
// WMMA GEMM: C[M,N] f32 = A[M,Kp]bf16 @ W[N,Kp]bf16^T (+bias), batched.
// Block = 8 waves; wave tile = 32(M) x 64(N); block tile = 256 x 64.
//
// The shared 64-col B panel is staged into LDS once per block per K-step via
// async global->LDS copies (one global_load_async_to_lds_b128 per wave per
// step: 256 lanes x 16B == 64 cols x 32 k x bf16), double buffered.
// The LDS destination address passed to the asm is derived from &shB so the
// object escapes and the compiler cannot fold the post-barrier ds reads.
// LDS layout is the WMMA B-fragment order [tile][khalf][q][lane16] so reads
// are bank-conflict-free ds_load_b128. A-frags stay as direct global loads.
// ---------------------------------------------------------------------------
__global__ __launch_bounds__(256) void k_gemm(
    const uint16_t* __restrict__ A, long long sA, int lda,
    const uint16_t* __restrict__ W, long long sW, int ldw,
    const float* __restrict__ bias,
    float* __restrict__ C, long long sC, int ldc,
    int M, int N, int Kp) {
  __shared__ uint16_t shB[2][2048];            // 2 x 4KB B-tile buffers

  const int tid  = threadIdx.x;
  const int wave = tid >> 5;
  const int lane = tid & 31;
  const int half = lane >> 4;
  const int l16  = lane & 15;

  A += (long long)blockIdx.z * sA;
  W += (long long)blockIdx.z * sW;
  C += (long long)blockIdx.z * sC;

  const int m0 = blockIdx.x * 256 + wave * 32;
  const int n0 = blockIdx.y * 64;

  int ar0 = m0 + l16;      if (ar0 >= M) ar0 = M - 1;
  int ar1 = m0 + 16 + l16; if (ar1 >= M) ar1 = M - 1;
  const uint16_t* A0 = A + (long long)ar0 * lda;
  const uint16_t* A1 = A + (long long)ar1 * lda;

  // ---- async staging mapping: tid -> (col, khalf, 16B-chunk) ----
  const int sc    = tid & 63;                  // column within panel
  const int shalf = (tid >> 6) & 1;            // k half (0..1)
  const int sq    = tid >> 7;                  // 16B chunk (0..1)
  int scol = n0 + sc; if (scol >= N) scol = N - 1;
  const uint16_t* gsrc = W + (long long)scol * ldw + sq * 16 + shalf * 8;
  // LDS byte address of this thread's 16B staging slot (escapes shB).
  const uint32_t lds_base = (uint32_t)(uintptr_t)(&shB[0][0]);
  const uint32_t lds_off = lds_base +
      (uint32_t)(((((sc >> 4) * 2 + shalf) * 2 + sq) * 16 + (sc & 15)) * 16);

  v8f acc[2][4];
#pragma unroll
  for (int t = 0; t < 4; ++t) { acc[0][t] = zero8(); acc[1][t] = zero8(); }

  // prologue: stage k0 = 0 into buffer 0
  {
    unsigned long long ga = (unsigned long long)gsrc;
    asm volatile("global_load_async_to_lds_b128 %0, %1, off"
                 :: "v"(lds_off), "v"(ga) : "memory");
  }

  int buf = 0;
  for (int k0 = 0; k0 < Kp; k0 += 32) {
    if (k0 + 32 < Kp) {                        // prefetch next tile (uniform)
      unsigned long long ga = (unsigned long long)(gsrc + k0 + 32);
      asm volatile("global_load_async_to_lds_b128 %0, %1, off"
                   :: "v"(lds_off + (uint32_t)((buf ^ 1) << 12)), "v"(ga)
                   : "memory");
      asm volatile("s_wait_asynccnt 0x1" ::: "memory");
    } else {
      asm volatile("s_wait_asynccnt 0x0" ::: "memory");
    }
    __syncthreads();                           // current buffer visible to all

    FragBF a0, a1;
    a0.q[0] = *(const uint4*)(A0 + k0 + half * 8);
    a0.q[1] = *(const uint4*)(A0 + k0 + 16 + half * 8);
    a1.q[0] = *(const uint4*)(A1 + k0 + half * 8);
    a1.q[1] = *(const uint4*)(A1 + k0 + 16 + half * 8);

    const uint16_t* sb = &shB[buf][0];
#pragma unroll
    for (int t = 0; t < 4; ++t) {
      FragBF b;
      b.q[0] = *(const uint4*)(sb + ((t * 2 + half) * 2 + 0) * 128 + l16 * 8);
      b.q[1] = *(const uint4*)(sb + ((t * 2 + half) * 2 + 1) * 128 + l16 * 8);
      acc[0][t] = __builtin_amdgcn_wmma_f32_16x16x32_bf16(
          false, a0.v, false, b.v, (short)0, acc[0][t], false, false);
      acc[1][t] = __builtin_amdgcn_wmma_f32_16x16x32_bf16(
          false, a1.v, false, b.v, (short)0, acc[1][t], false, false);
    }
    __syncthreads();                           // reads done before overwrite
    buf ^= 1;
  }

  const int rb0 = m0 + half * 8;
  const int rb1 = m0 + 16 + half * 8;
#pragma unroll
  for (int t = 0; t < 4; ++t) {
    int col = n0 + t * 16 + l16;
    if (col >= N) continue;
    float bv = bias ? bias[col] : 0.0f;
#pragma unroll
    for (int r = 0; r < 8; ++r) {
      int row0 = rb0 + r;
      if (row0 < M) C[(long long)row0 * ldc + col] = acc[0][t][r] + bv;
      int row1 = rb1 + r;
      if (row1 < M) C[(long long)row1 * ldc + col] = acc[1][t][r] + bv;
    }
  }
}

// ---------------------------------------------------------------------------
// Per-column mean / rsqrt(var+eps) over M rows (BatchNorm batch stats)
// ---------------------------------------------------------------------------
__global__ __launch_bounds__(256) void k_colstats(const float* __restrict__ Y, int N,
                                                  int M, float* __restrict__ mean,
                                                  float* __restrict__ rsig) {
  const int col = blockIdx.x;
  float s = 0.0f, s2 = 0.0f;
  for (int r = threadIdx.x; r < M; r += 256) {
    float v = Y[(long long)r * N + col];
    s += v; s2 += v * v;
  }
  __shared__ float sh[512];
  sh[threadIdx.x] = s; sh[256 + threadIdx.x] = s2;
  __syncthreads();
  for (int off = 128; off > 0; off >>= 1) {
    if (threadIdx.x < (unsigned)off) {
      sh[threadIdx.x] += sh[threadIdx.x + off];
      sh[256 + threadIdx.x] += sh[256 + threadIdx.x + off];
    }
    __syncthreads();
  }
  if (threadIdx.x == 0) {
    float m = sh[0] / (float)M;
    float v = sh[256] / (float)M - m * m;
    mean[col] = m;
    rsig[col] = rsqrtf(v + 1e-5f);
  }
}

// ---------------------------------------------------------------------------
// BN (+GELU) apply, optional f32 out and zero-padded bf16 out
// ---------------------------------------------------------------------------
__global__ __launch_bounds__(256) void k_bnact(const float* __restrict__ Y, int N,
                                               const float* __restrict__ mean,
                                               const float* __restrict__ rsig,
                                               const float* __restrict__ g,
                                               const float* __restrict__ be,
                                               int do_gelu, float* __restrict__ outF,
                                               uint16_t* __restrict__ outB, int ldb,
                                               long long total) {
  long long idx = (long long)blockIdx.x * blockDim.x + threadIdx.x;
  if (idx >= total) return;
  long long row = idx / ldb;
  int c = (int)(idx % ldb);
  if (c < N) {
    float v = (Y[row * N + c] - mean[c]) * rsig[c] * g[c] + be[c];
    if (do_gelu) v = gelu_exact(v);
    if (outF) outF[row * N + c] = v;
    if (outB) outB[row * (long long)ldb + c] = f2b(v);
  } else if (outB) {
    outB[row * (long long)ldb + c] = 0;
  }
}

// ---------------------------------------------------------------------------
// Elementwise: out = gelu?(A + R), optional f32 out and padded bf16 out
// ---------------------------------------------------------------------------
__global__ __launch_bounds__(256) void k_ewise(const float* __restrict__ A,
                                               const float* __restrict__ R, int N,
                                               int do_gelu, float* __restrict__ outF,
                                               uint16_t* __restrict__ outB, int ldb,
                                               long long total) {
  long long idx = (long long)blockIdx.x * blockDim.x + threadIdx.x;
  if (idx >= total) return;
  long long row = idx / ldb;
  int c = (int)(idx % ldb);
  if (c < N) {
    float v = A[row * N + c];
    if (R) v += R[row * N + c];
    if (do_gelu) v = gelu_exact(v);
    if (outF) outF[row * N + c] = v;
    if (outB) outB[row * (long long)ldb + c] = f2b(v);
  } else if (outB) {
    outB[row * (long long)ldb + c] = 0;
  }
}

// ---------------------------------------------------------------------------
// h = feats + pose (broadcast over batch); f32 + bf16 outputs
// ---------------------------------------------------------------------------
__global__ __launch_bounds__(256) void k_addpose(const float* __restrict__ F,
                                                 const float* __restrict__ pose,
                                                 float* __restrict__ Hf,
                                                 uint16_t* __restrict__ Hb,
                                                 long long total) {
  long long idx = (long long)blockIdx.x * blockDim.x + threadIdx.x;
  if (idx >= total) return;
  int c = (int)(idx % CDIM);
  long long row = idx / CDIM;
  int n = (int)(row & 1023);
  float v = F[idx] + pose[n * CDIM + c];
  Hf[idx] = v;
  Hb[idx] = f2b(v);
}

// ---------------------------------------------------------------------------
// Top-k over 1024-wide similarity rows; one wave32 per row, row in registers
// ---------------------------------------------------------------------------
__global__ __launch_bounds__(128) void k_topk(const float* __restrict__ sim,
                                              int* __restrict__ graph, int k) {
  const int lane = threadIdx.x & 31;
  const long long row = (long long)blockIdx.x * 4 + (threadIdx.x >> 5);
  const float* S = sim + row * NPATCH;
  float vals[32];
#pragma unroll
  for (int j = 0; j < 32; ++j) vals[j] = S[lane + 32 * j];
  for (int it = 0; it < k; ++it) {
    float bv = -3.402823466e38f;
    int bi = 0;
#pragma unroll
    for (int j = 0; j < 32; ++j) {
      if (vals[j] > bv) { bv = vals[j]; bi = lane + 32 * j; }
    }
    for (int off = 16; off > 0; off >>= 1) {
      float ov = __shfl_xor(bv, off, 32);
      int   oi = __shfl_xor(bi, off, 32);
      if (ov > bv || (ov == bv && oi < bi)) { bv = ov; bi = oi; }
    }
    if (lane == 0) graph[row * k + it] = bi;
    if ((bi & 31) == lane) vals[bi >> 5] = -3.402823466e38f;
  }
}

// ---------------------------------------------------------------------------
// Max-relative aggregate + interleaved stack -> bf16 [M, 2C]
// ---------------------------------------------------------------------------
__global__ __launch_bounds__(256) void k_maxrel(const float* __restrict__ X,
                                                const int* __restrict__ graph, int k,
                                                uint16_t* __restrict__ stacked,
                                                long long total) {
  long long idx = (long long)blockIdx.x * blockDim.x + threadIdx.x;
  if (idx >= total) return;
  int c = (int)(idx % CDIM);
  long long row = idx / CDIM;
  long long bbase = (row >> 10) << 10;         // batch * 1024
  float xv = X[row * CDIM + c];
  float m = -3.402823466e38f;
  const int* g = graph + row * k;
  for (int j = 0; j < k; ++j) {
    int nb = g[j];
    float v = X[(bbase + nb) * CDIM + c];
    m = fmaxf(m, v);
  }
  stacked[row * (2 * CDIM) + 2 * c]     = f2b(xv);
  stacked[row * (2 * CDIM) + 2 * c + 1] = f2b(m - xv);
}

// ---------------------------------------------------------------------------
// TDM probe (compiled into bundle, never launched): keeps the
// tensor_load_to_lds / s_wait_tensorcnt path validated for future use.
// ---------------------------------------------------------------------------
__global__ void k_tdm_probe(uint32_t* __restrict__ out) {
  __shared__ uint32_t sh[256];
  u32x4 g0 = {};
  i32x8 g1 = {};
  i32x4 g2 = {};
  i32x4 g3 = {};
#if defined(__clang_major__) && __clang_major__ >= 23
  i32x8 g4 = {};
  __builtin_amdgcn_tensor_load_to_lds(g0, g1, g2, g3, g4, 0);
#else
  __builtin_amdgcn_tensor_load_to_lds(g0, g1, g2, g3, 0);
#endif
  __builtin_amdgcn_s_wait_tensorcnt((short)0);
  __syncthreads();
  out[threadIdx.x] = sh[threadIdx.x];
}

// ---------------------------------------------------------------------------
// Host orchestration
// ---------------------------------------------------------------------------
static inline int pad32(int v) { return (v + 31) & ~31; }

extern "C" void kernel_launch(void* const* d_in, const int* in_sizes, int n_in,
                              void* d_out, int out_size, void* d_ws, size_t ws_size,
                              hipStream_t stream) {
  (void)in_sizes; (void)out_size; (void)ws_size;

  // ---- input pointers (setup_inputs insertion order: x, then params) ----
  int p = 0;
  auto nxt = [&]() -> const float* {
    const float* q = (const float*)d_in[p < n_in ? p : 0];
    ++p;
    return q;
  };
  const float* x = nxt();
  const float *ew[6], *eb[6], *eg[6], *ebe[6];
  for (int i = 0; i < 6; ++i) { ew[i] = nxt(); eb[i] = nxt(); eg[i] = nxt(); ebe[i] = nxt(); }
  const float* pose = nxt();
  struct TLp { const float *w1, *b1, *g, *be, *w2, *b2; };
  struct Bp { TLp in1, out1, in2, out2; const float *fcw, *fcb; } Bk[4];
  auto tl = [&]() { TLp t; t.w1 = nxt(); t.b1 = nxt(); t.g = nxt(); t.be = nxt(); t.w2 = nxt(); t.b2 = nxt(); return t; };
  for (int i = 0; i < 4; ++i) {
    Bk[i].in1 = tl(); Bk[i].out1 = tl(); Bk[i].in2 = tl(); Bk[i].out2 = tl();
    Bk[i].fcw = nxt(); Bk[i].fcb = nxt();
  }

  // ---- workspace bump allocator ----
  char* ws = (char*)d_ws;
  size_t off = 0;
  auto alloc = [&](size_t bytes) -> void* {
    void* q = ws + off;
    off = (off + bytes + 255) & ~(size_t)255;
    return q;
  };
  uint16_t* Ab0 = (uint16_t*)alloc(MROWS * 768 * 2);
  uint16_t* Ab1 = (uint16_t*)alloc(MROWS * 768 * 2);
  float*    Y0  = (float*)alloc(MROWS * 768 * 4);
  float*    Y1  = (float*)alloc(MROWS * 768 * 4);
  float*    Hf  = (float*)alloc(MROWS * CDIM * 4);
  uint16_t* Hb  = (uint16_t*)alloc(MROWS * CDIM * 2);
  float*    X1f = (float*)alloc(MROWS * CDIM * 4);
  float*    X4f = (float*)alloc(MROWS * CDIM * 4);
  float*    simB = (float*)alloc(32LL * NPATCH * NPATCH * 4);
  int*      graph = (int*)alloc(32LL * NPATCH * 18 * 4);
  float*    mean = (float*)alloc(1024 * 4);
  float*    rsig = (float*)alloc(1024 * 4);

  // ---- helpers ----
  auto cvtW = [&](const float* src, int N, int K) -> uint16_t* {
    int Kp = pad32(K);
    uint16_t* dst = (uint16_t*)alloc((size_t)N * Kp * 2);
    long long total = (long long)N * Kp;
    k_w2b<<<(unsigned)((total + 255) / 256), 256, 0, stream>>>(src, N, K, Kp, dst, total);
    return dst;
  };
  auto gemm = [&](const uint16_t* A, int lda, const uint16_t* W, int ldw,
                  const float* bias, float* C, int ldc, int M, int N, int Kp,
                  int batch, long long sA, long long sW, long long sC) {
    dim3 g((M + 255) / 256, (N + 63) / 64, batch);
    k_gemm<<<g, 256, 0, stream>>>(A, sA, lda, W, sW, ldw, bias, C, sC, ldc, M, N, Kp);
  };
  auto stats = [&](const float* Y, int N) {
    k_colstats<<<N, 256, 0, stream>>>(Y, N, (int)MROWS, mean, rsig);
  };
  auto bnact = [&](const float* Y, int N, const float* g, const float* be, int gl,
                   float* oF, uint16_t* oB, int ldb) {
    long long total = MROWS * (long long)ldb;
    k_bnact<<<(unsigned)((total + 255) / 256), 256, 0, stream>>>(Y, N, mean, rsig, g, be,
                                                                 gl, oF, oB, ldb, total);
  };
  auto ewise = [&](const float* A, const float* R, int N, int gl, float* oF,
                   uint16_t* oB, int ldb) {
    long long total = MROWS * (long long)ldb;
    k_ewise<<<(unsigned)((total + 255) / 256), 256, 0, stream>>>(A, R, N, gl, oF, oB, ldb, total);
  };

  // ---- stage 1: patchify ----
  k_patchify<<<(unsigned)((MROWS * 768 + 255) / 256), 256, 0, stream>>>(x, Ab0);

  // ---- stage 2: embed MLP 768->96->48->24->48->96->192 ----
  const int edin[6]  = {768, 96, 48, 24, 48, 96};
  const int edout[6] = {96, 48, 24, 48, 96, 192};
  uint16_t* curA = Ab0;
  uint16_t* nxtA = Ab1;
  for (int i = 0; i < 6; ++i) {
    int Kp = pad32(edin[i]), N = edout[i], KpN = pad32(N);
    uint16_t* Wb = cvtW(ew[i], N, edin[i]);
    gemm(curA, Kp, Wb, Kp, eb[i], Y0, N, (int)MROWS, N, Kp, 1, 0, 0, 0);
    stats(Y0, N);
    if (i < 5) {
      bnact(Y0, N, eg[i], ebe[i], 1, nullptr, nxtA, KpN);
      uint16_t* t = curA; curA = nxtA; nxtA = t;
    } else {
      bnact(Y0, N, eg[i], ebe[i], 0, Y1, nullptr, N);
    }
  }
  k_addpose<<<(unsigned)((MROWS * CDIM + 255) / 256), 256, 0, stream>>>(Y1, pose, Hf, Hb,
                                                                        MROWS * CDIM);

  // ---- stage 3: 4 ViG blocks ----
  for (int blk = 0; blk < 4; ++blk) {
    const int kk = 9 + blk;
    const Bp& B = Bk[blk];
    uint16_t* Win1a = cvtW(B.in1.w1, 192, 192);
    uint16_t* Win1b = cvtW(B.in1.w2, 192, 192);
    uint16_t* Wout1a = cvtW(B.out1.w1, 192, 192);
    uint16_t* Wout1b = cvtW(B.out1.w2, 192, 192);
    uint16_t* Win2a = cvtW(B.in2.w1, 768, 192);
    uint16_t* Win2b = cvtW(B.in2.w2, 192, 768);
    uint16_t* Wout2a = cvtW(B.out2.w1, 768, 192);
    uint16_t* Wout2b = cvtW(B.out2.w2, 192, 768);
    uint16_t* Wfc = cvtW(B.fcw, 192, 384);

    // sim = h h^T  (batched 1024x1024x192), then top-k
    gemm(Hb, CDIM, Hb, CDIM, nullptr, simB, NPATCH, NPATCH, NPATCH, CDIM, 32,
         (long long)NPATCH * CDIM, (long long)NPATCH * CDIM,
         (long long)NPATCH * NPATCH);
    k_topk<<<32 * NPATCH / 4, 128, 0, stream>>>(simB, graph, kk);

    // in1 two_layer on h
    gemm(Hb, CDIM, Win1a, CDIM, B.in1.b1, Y0, CDIM, (int)MROWS, CDIM, CDIM, 1, 0, 0, 0);
    stats(Y0, CDIM);
    bnact(Y0, CDIM, B.in1.g, B.in1.be, 1, nullptr, Ab0, CDIM);
    gemm(Ab0, CDIM, Win1b, CDIM, B.in1.b2, Y1, CDIM, (int)MROWS, CDIM, CDIM, 1, 0, 0, 0);
    ewise(Y1, Hf, CDIM, 0, X1f, nullptr, CDIM);                 // x1

    // graph aggregate -> stacked [M, 384] bf16
    k_maxrel<<<(unsigned)((MROWS * CDIM + 255) / 256), 256, 0, stream>>>(
        X1f, graph, kk, Ab1, MROWS * CDIM);

    // fc: 384 -> 192, then gelu(x2)
    gemm(Ab1, 384, Wfc, 384, B.fcb, Y0, CDIM, (int)MROWS, CDIM, 384, 1, 0, 0, 0);
    ewise(Y0, nullptr, CDIM, 1, Y1, Ab0, CDIM);                 // g2 (f32 Y1, bf16 Ab0)

    // out1 two_layer on g2
    gemm(Ab0, CDIM, Wout1a, CDIM, B.out1.b1, Y0, CDIM, (int)MROWS, CDIM, CDIM, 1, 0, 0, 0);
    stats(Y0, CDIM);
    bnact(Y0, CDIM, B.out1.g, B.out1.be, 1, nullptr, Ab1, CDIM);
    gemm(Ab1, CDIM, Wout1b, CDIM, B.out1.b2, Y0, CDIM, (int)MROWS, CDIM, CDIM, 1, 0, 0, 0);
    ewise(Y0, Y1, CDIM, 0, Y0, nullptr, CDIM);                  // x3 = .. + g2
    ewise(Y0, Hf, CDIM, 0, X4f, Hb, CDIM);                      // x4 (bf16 into Hb)

    // in2 two_layer: 192 -> 768 -> 192
    gemm(Hb, CDIM, Win2a, CDIM, B.in2.b1, Y0, 768, (int)MROWS, 768, CDIM, 1, 0, 0, 0);
    stats(Y0, 768);
    bnact(Y0, 768, B.in2.g, B.in2.be, 1, nullptr, Ab0, 768);
    gemm(Ab0, 768, Win2b, 768, B.in2.b2, Y1, CDIM, (int)MROWS, CDIM, 768, 1, 0, 0, 0);
    ewise(Y1, X4f, CDIM, 1, Y0, Ab1, CDIM);                     // g5 = gelu(x5)

    // out2 two_layer on g5
    gemm(Ab1, CDIM, Wout2a, CDIM, B.out2.b1, Y1, 768, (int)MROWS, 768, CDIM, 1, 0, 0, 0);
    stats(Y1, 768);
    bnact(Y1, 768, B.out2.g, B.out2.be, 1, nullptr, Ab0, 768);
    gemm(Ab0, 768, Wout2b, 768, B.out2.b2, Y1, CDIM, (int)MROWS, CDIM, 768, 1, 0, 0, 0);
    ewise(Y1, Y0, CDIM, 0, Y1, nullptr, CDIM);                  // x6 = .. + g5

    const bool last = (blk == 3);
    ewise(Y1, X4f, CDIM, 0, last ? (float*)d_out : Hf, last ? nullptr : Hb, CDIM);
  }
}
